// CommNetWork_Critic_23081154249051
// MI455X (gfx1250) — matrix operand order repaired and checked by reading, not compiled, not run on hardware
//
#include <hip/hip_runtime.h>
#include <hip/hip_bf16.h>

// ---------------------------------------------------------------------------
// CommNet critic, fully fused per 128-row tile (4 batches x 32 agents).
// bf16 WMMA (v_wmma_f32_16x16x32_bf16), f32 accumulate, in-place GRU updates
// so only 2x64KB LDS buffers are needed -> 2 workgroups / WGP.
// ---------------------------------------------------------------------------

typedef __attribute__((ext_vector_type(8)))  float   v8f;
typedef __attribute__((ext_vector_type(4)))  __bf16  v4bf;
typedef __attribute__((ext_vector_type(8)))  __bf16  v8bf;
typedef __attribute__((ext_vector_type(16))) __bf16  v16bf;

union V16 { v16bf v; v8bf h[2]; };

#define DEV __device__ __forceinline__

DEV float sig_(float x)  { return 1.0f / (1.0f + __expf(-x)); }
DEV float tanh_(float x) { float e = __expf(-2.0f * x); return (1.0f - e) / (1.0f + e); }

// Preload one 16xK A-tile (row-major LDS, lda elems/row) into registers.
// Lane layout per ISA 7.12.2 (16-bit A 16x32 per 32-K chunk).
template <int KC>
DEV void preload_A(const __bf16* A, int lda, V16 a[KC]) {
  const int lane = threadIdx.x & 31;
  const __bf16* aRow = A + (lane & 15) * lda + (lane >> 4) * 8;
#pragma unroll
  for (int c = 0; c < KC; ++c) {
    a[c].h[0] = *(const v8bf*)(aRow + c * 32);
    a[c].h[1] = *(const v8bf*)(aRow + c * 32 + 16);
  }
}

// 16x16 tile of A[16,K] @ W[N,K]^T with A in registers, W row-major global.
template <int KC>
DEV v8f gemm_regA(const V16 a[KC], const __bf16* __restrict__ W, int ldw, v8f acc) {
  const int lane = threadIdx.x & 31;
  const __bf16* wRow = W + (lane & 15) * ldw + (lane >> 4) * 8;
#pragma unroll
  for (int c = 0; c < KC; ++c) {
    V16 b;
    b.h[0] = *(const v8bf*)(wRow + c * 32);
    b.h[1] = *(const v8bf*)(wRow + c * 32 + 16);
    acc = __builtin_amdgcn_wmma_f32_16x16x32_bf16(
        false, a[c].v, false, b.v, (short)0, acc, false, false);
  }
  return acc;
}

__global__ void cvt_bf16_kernel(const float* __restrict__ src,
                                __bf16* __restrict__ dst, int n) {
  int i = blockIdx.x * blockDim.x + threadIdx.x;
  if (i < n) dst[i] = (__bf16)src[i];
}

__global__ __launch_bounds__(256)
void commnet_fused_kernel(const float*  __restrict__ obs,
                          const __bf16* __restrict__ wEnc,   // [256,128]
                          const __bf16* __restrict__ wFobs,  // [256,256]
                          const __bf16* __restrict__ wIh,    // [768,256]
                          const __bf16* __restrict__ wHh,    // [768,256]
                          const float*  __restrict__ enc_b,
                          const float*  __restrict__ fobs_b,
                          const float*  __restrict__ b_ih,
                          const float*  __restrict__ b_hh,
                          const float*  __restrict__ dec_W,
                          const float*  __restrict__ dec_b,
                          float* __restrict__ out) {
  constexpr int H = 256;
  extern __shared__ char smem[];
  __bf16* P    = (__bf16*)(smem);             // 64KB: X(32KB) -> Hf -> h1 -> h2
  __bf16* Q    = (__bf16*)(smem + 65536);     // 64KB: E -> c
  float*  scol = (float*) (smem + 131072);    // 4KB : col sums / dec partials

  const int tid   = threadIdx.x;
  const int wave  = tid >> 5;                 // == m-tile owned by this wave
  const int lane  = tid & 31;
  const int col16 = lane & 15;
  const int hi    = lane >> 4;
  const int bt    = blockIdx.x;               // 128-row tile index
  const int m     = wave;

  // ---- Stage 1: obs tile f32 -> bf16 into P as X[128][128] ----
  {
    const float4* src = (const float4*)(obs + (size_t)bt * (128 * 128));
    for (int i = tid; i < (128 * 128) / 4; i += 256) {
      float4 v = src[i];
      __bf16* d = P + i * 4;
      d[0] = (__bf16)v.x; d[1] = (__bf16)v.y;
      d[2] = (__bf16)v.z; d[3] = (__bf16)v.w;
    }
  }
  __syncthreads();

  // ---- Stage 2: E = relu(X @ encW^T + enc_b) -> Q [128][256] ----
  {
    V16 ax[4];
    preload_A<4>(P + m * 16 * 128, 128, ax);
#pragma unroll 1
    for (int n = 0; n < 16; ++n) {
      v8f acc = {};
      acc = gemm_regA<4>(ax, wEnc + n * 16 * 128, 128, acc);
      int col = n * 16 + col16;
      float bv = enc_b[col];
#pragma unroll
      for (int i = 0; i < 8; ++i) {
        int row = m * 16 + i + hi * 8;
        Q[row * H + col] = (__bf16)fmaxf(acc[i] + bv, 0.0f);
      }
    }
  }
  __syncthreads();

  // ---- Stage 3: Hf = E @ fobsW^T + fobs_b -> P (overwrites X) ----
  {
    V16 ae[8];
    preload_A<8>(Q + m * 16 * H, H, ae);
#pragma unroll 1
    for (int n = 0; n < 16; ++n) {
      v8f acc = {};
      acc = gemm_regA<8>(ae, wFobs + n * 16 * H, H, acc);
      int col = n * 16 + col16;
      float bv = fobs_b[col];
#pragma unroll
      for (int i = 0; i < 8; ++i) {
        int row = m * 16 + i + hi * 8;
        P[row * H + col] = (__bf16)(acc[i] + bv);
      }
    }
  }
  __syncthreads();

  // ---- Stage 4: GRU #1 (x=0): h1 = GRU(0, Hf), in place in P ----
  {
    V16 ah[8];                                   // Hf tile in registers
    preload_A<8>(P + m * 16 * H, H, ah);         // loaded before any store
#pragma unroll 1
    for (int n = 0; n < 16; ++n) {
      v8f ar = {}, az = {}, an = {};
      ar = gemm_regA<8>(ah, wHh + (0   + n * 16) * H, H, ar);
      az = gemm_regA<8>(ah, wHh + (256 + n * 16) * H, H, az);
      an = gemm_regA<8>(ah, wHh + (512 + n * 16) * H, H, an);
      int col = n * 16 + col16;
      float bihr = b_ih[col], bihz = b_ih[256 + col], bihn = b_ih[512 + col];
      float bhhr = b_hh[col], bhhz = b_hh[256 + col], bhhn = b_hh[512 + col];
#pragma unroll
      for (int i = 0; i < 8; ++i) {
        int row = m * 16 + i + hi * 8;
        float rr = sig_(bihr + ar[i] + bhhr);
        float zz = sig_(bihz + az[i] + bhhz);
        float nn = tanh_(bihn + rr * (an[i] + bhhn));
        float hf = (float)P[row * H + col];      // read-before-write, same lane
        P[row * H + col] = (__bf16)((1.0f - zz) * nn + zz * hf);
      }
    }
  }
  __syncthreads();

  // ---- Stage 5: comm  c = (sum_agents h1 - h1)/32 -> Q ----
  {
    int g  = tid >> 6;                // batch group 0..3
    int c4 = (tid & 63) * 4;          // 4 columns per thread
    const __bf16* base = P + (g * 32) * H + c4;
    float s0 = 0.f, s1 = 0.f, s2 = 0.f, s3 = 0.f;
#pragma unroll 8
    for (int j = 0; j < 32; ++j) {
      v4bf v = *(const v4bf*)(base + j * H);
      s0 += (float)v[0]; s1 += (float)v[1]; s2 += (float)v[2]; s3 += (float)v[3];
    }
    *(float4*)(scol + g * H + c4) = make_float4(s0, s1, s2, s3);
  }
  __syncthreads();
  for (int i = tid; i < (128 * H) / 4; i += 256) {
    int row = i >> 6;
    int c4  = (i & 63) * 4;
    int g   = row >> 5;
    v4bf h  = *(const v4bf*)(P + row * H + c4);
    float4 sc = *(const float4*)(scol + g * H + c4);
    v4bf o;
    o[0] = (__bf16)((sc.x - (float)h[0]) * (1.0f / 32.0f));
    o[1] = (__bf16)((sc.y - (float)h[1]) * (1.0f / 32.0f));
    o[2] = (__bf16)((sc.z - (float)h[2]) * (1.0f / 32.0f));
    o[3] = (__bf16)((sc.w - (float)h[3]) * (1.0f / 32.0f));
    *(v4bf*)(Q + row * H + c4) = o;
  }
  __syncthreads();

  // ---- Stage 6: GRU #2: h2 = GRU(c, h1), in place in P ----
  {
    V16 ah[8], ac[8];
    preload_A<8>(P + m * 16 * H, H, ah);         // h1, before any store to P
    preload_A<8>(Q + m * 16 * H, H, ac);         // c
#pragma unroll 1
    for (int n = 0; n < 16; ++n) {
      v8f gir = {}, giz = {}, gin = {}, ghr = {}, ghz = {}, ghn = {};
      gir = gemm_regA<8>(ac, wIh + (0   + n * 16) * H, H, gir);
      giz = gemm_regA<8>(ac, wIh + (256 + n * 16) * H, H, giz);
      gin = gemm_regA<8>(ac, wIh + (512 + n * 16) * H, H, gin);
      ghr = gemm_regA<8>(ah, wHh + (0   + n * 16) * H, H, ghr);
      ghz = gemm_regA<8>(ah, wHh + (256 + n * 16) * H, H, ghz);
      ghn = gemm_regA<8>(ah, wHh + (512 + n * 16) * H, H, ghn);
      int col = n * 16 + col16;
      float bihr = b_ih[col], bihz = b_ih[256 + col], bihn = b_ih[512 + col];
      float bhhr = b_hh[col], bhhz = b_hh[256 + col], bhhn = b_hh[512 + col];
#pragma unroll
      for (int i = 0; i < 8; ++i) {
        int row = m * 16 + i + hi * 8;
        float rr = sig_(gir[i] + bihr + ghr[i] + bhhr);
        float zz = sig_(giz[i] + bihz + ghz[i] + bhhz);
        float nn = tanh_(gin[i] + bihn + rr * (ghn[i] + bhhn));
        float h1 = (float)P[row * H + col];      // read-before-write, same lane
        P[row * H + col] = (__bf16)((1.0f - zz) * nn + zz * h1);
      }
    }
  }
  __syncthreads();

  // ---- Stage 7: out = h2 @ decW^T + dec_b (2 threads per row) ----
  {
    int row  = tid >> 1;
    int half = tid & 1;
    const __bf16* hrow = P + row * H + half * 128;
    const float*  wrow = dec_W + half * 128;
    float s = 0.0f;
#pragma unroll
    for (int c = 0; c < 128; c += 8) {
      v8bf v = *(const v8bf*)(hrow + c);
#pragma unroll
      for (int j = 0; j < 8; ++j) s += (float)v[j] * wrow[c + j];
    }
    scol[tid] = s;
  }
  __syncthreads();
  if (tid < 128) {
    out[(size_t)bt * 128 + tid] = scol[2 * tid] + scol[2 * tid + 1] + dec_b[0];
  }
}

extern "C" void kernel_launch(void* const* d_in, const int* in_sizes, int n_in,
                              void* d_out, int out_size, void* d_ws, size_t ws_size,
                              hipStream_t stream) {
  const float* obs    = (const float*)d_in[0];
  // d_in[1] = act, unused by the reference
  const float* enc_W  = (const float*)d_in[2];
  const float* enc_b  = (const float*)d_in[3];
  const float* fobs_W = (const float*)d_in[4];
  const float* fobs_b = (const float*)d_in[5];
  const float* W_ih   = (const float*)d_in[6];
  const float* b_ih   = (const float*)d_in[7];
  const float* W_hh   = (const float*)d_in[8];
  const float* b_hh   = (const float*)d_in[9];
  const float* dec_W  = (const float*)d_in[10];
  const float* dec_b  = (const float*)d_in[11];
  float* out = (float*)d_out;

  // bf16 weight copies in workspace (L2-resident, ~1 MB total)
  char* ws = (char*)d_ws;
  __bf16* wEnc  = (__bf16*)(ws);            // 32768  elems
  __bf16* wFobs = (__bf16*)(ws + 65536);    // 65536  elems
  __bf16* wIh   = (__bf16*)(ws + 196608);   // 196608 elems
  __bf16* wHh   = (__bf16*)(ws + 589824);   // 196608 elems

  cvt_bf16_kernel<<<(32768  + 255) / 256, 256, 0, stream>>>(enc_W,  wEnc,  32768);
  cvt_bf16_kernel<<<(65536  + 255) / 256, 256, 0, stream>>>(fobs_W, wFobs, 65536);
  cvt_bf16_kernel<<<(196608 + 255) / 256, 256, 0, stream>>>(W_ih,   wIh,   196608);
  cvt_bf16_kernel<<<(196608 + 255) / 256, 256, 0, stream>>>(W_hh,   wHh,   196608);

  // 65536 rows / 128 rows-per-tile = 512 workgroups
  // LDS: 2x64KB buffers + 4KB scratch = 132KB -> 2 workgroups per WGP
  const size_t ldsBytes = 2 * 65536 + 4096;
  commnet_fused_kernel<<<512, 256, ldsBytes, stream>>>(
      obs, wEnc, wFobs, wIh, wHh,
      enc_b, fobs_b, b_ih, b_hh, dec_W, dec_b, out);
}
